// two_layer_model_11622181503322
// MI455X (gfx1250) — matrix operand order repaired and checked by reading, compiled, not run on hardware
//
#include <hip/hip_runtime.h>
#include <hip/hip_bf16.h>

#define BB 32
#define VV 20
#define NDC 40
#define NRC 40
#define DXV_ 10000
#define DRV_ 5000
#define NANC_ 2000
#define SPLIT_ 4
#define DD 128
#define PATTN_ 128
#define LBL_ 2000
#define EE 200000
#define NN_ 18000   /* DXV + SPLIT*NANC */
#define NEG_SLOPE 0.2f

typedef __attribute__((ext_vector_type(16))) __bf16 bf16x16;
typedef __attribute__((ext_vector_type(8)))  float  f32x8;

// ---------- helpers ----------
__device__ inline unsigned short f2bf(float x) {
    union { float f; unsigned u; } v; v.f = x;
    unsigned r = v.u + 0x7FFFu + ((v.u >> 16) & 1u);
    return (unsigned short)(r >> 16);
}
__device__ inline unsigned pack2(float a, float b) {
    return (unsigned)f2bf(a) | ((unsigned)f2bf(b) << 16);
}
__device__ inline float bfbits2f(unsigned hs) {
    union { unsigned u; float f; } v; v.u = hs << 16; return v.f;
}

// A fragment (16x32 bf16) from LDS stored as u32 pairs, row stride stride_u32,
// starting at pair offset kpair0.  ISA layout: lanes0-15 m=lane, pairs j=0..3,8..11;
// lanes16-31 pairs j=4..7,12..15.
__device__ inline bf16x16 load_a_frag(const unsigned* lds, int m0, int stride_u32, int kpair0) {
    int lane = threadIdx.x & 31;
    int m = m0 + (lane & 15);
    int jb = (lane >> 4) ? 4 : 0;
    const unsigned* p = lds + m * stride_u32 + kpair0 + jb;
    union { bf16x16 v; unsigned u[8]; } r;
#pragma unroll
    for (int i = 0; i < 4; ++i) r.u[i] = p[i];        // 16B aligned -> ds_load_b128
#pragma unroll
    for (int i = 0; i < 4; ++i) r.u[4 + i] = p[8 + i];
    return r.v;
}

// B fragment (32x16 bf16) from pre-packed global: 32 lanes x 8 u32, contiguous.
__device__ inline bf16x16 load_b_frag(const unsigned* bp) {
    int lane = threadIdx.x & 31;
    const unsigned* p = bp + lane * 8;
    union { bf16x16 v; unsigned u[8]; } r;
#pragma unroll
    for (int i = 0; i < 8; ++i) r.u[i] = p[i];
    return r.v;
}

__device__ inline void atomicMaxFloat(float* addr, float v) {
    if (v >= 0.0f) atomicMax((int*)addr, __float_as_int(v));
    else           atomicMin((unsigned int*)addr, __float_as_uint(v));
}

// ---------- pack B (row-major fp32 KxN) into WMMA fragment layout ----------
__global__ void pack_b_kernel(const float* __restrict__ W, unsigned* __restrict__ Bp,
                              int K, int Ncols, int NT) {
    int tile = blockIdx.x;          // kt*NT + nt
    int lane = threadIdx.x;         // 32 threads
    int kt = tile / NT, nt = tile % NT;
    int n = nt * 16 + (lane & 15);
    int kb = kt * 32 + ((lane >> 4) ? 8 : 0);
    unsigned* dst = Bp + (size_t)tile * 256 + lane * 8;
#pragma unroll
    for (int i = 0; i < 8; ++i) {
        int k0 = kb + 2 * i + (i >= 4 ? 8 : 0);
        float v0 = (k0     < K && n < Ncols) ? W[(size_t)k0 * Ncols + n]       : 0.0f;
        float v1 = (k0 + 1 < K && n < Ncols) ? W[(size_t)(k0 + 1) * Ncols + n] : 0.0f;
        dst[i] = pack2(v0, v1);
    }
}

// ---------- init scratch ----------
__global__ void init_kernel(float* gat, float* mmax, float* denom) {
    int i = blockIdx.x * blockDim.x + threadIdx.x;
    if (i < DXV_ * DD) gat[i] = 0.0f;
    if (i < DXV_) { mmax[i] = -1e30f; denom[i] = 1e-9f; }
}

// ---------- generic GEMM: C[M,128] = A[M,K] @ Bpacked ; A rows split across two tables
__global__ void gemm128_kernel(const float* __restrict__ A0, const float* __restrict__ A1,
                               int splitRows, const unsigned* __restrict__ Bp,
                               float* __restrict__ C, int M, int K, int KT) {
    __shared__ unsigned alds[64 * 16];   // 64 rows x 32 bf16 (16 u32 pairs)
    const int tid = threadIdx.x;
    const int wave = tid >> 5, lane = tid & 31;
    const int rowb = blockIdx.x * 64;
    const bool rowFull = (rowb + 64 <= M);
    f32x8 acc[4];
#pragma unroll
    for (int t = 0; t < 4; ++t)
#pragma unroll
        for (int i = 0; i < 8; ++i) acc[t][i] = 0.0f;

    // per-thread staging coordinates (4 u32-pairs per thread per k-tile, idx step 256)
    for (int kt = 0; kt < KT; ++kt) {
        __syncthreads();
        const bool kFull = (kt * 32 + 32 <= K);
        if (rowFull && kFull) {
            // fast path: vector loads, no guards (block-uniform branch)
            for (int idx = tid; idx < 64 * 16; idx += 256) {
                int r = idx >> 4, jp = idx & 15;
                int g = rowb + r;
                const float* src = (g < splitRows) ? (A0 + (size_t)g * K)
                                                   : (A1 + (size_t)(g - splitRows) * K);
                const float2 v = *(const float2*)(src + kt * 32 + jp * 2);
                if (kt + 2 < KT)  // keep the HBM stream ahead (global_prefetch_b8)
                    __builtin_prefetch(src + (kt + 2) * 32 + jp * 2, 0, 0);
                alds[idx] = pack2(v.x, v.y);
            }
        } else {
            for (int idx = tid; idx < 64 * 16; idx += 256) {
                int r = idx >> 4, jp = idx & 15;
                int g = rowb + r;
                int k0 = kt * 32 + jp * 2;
                float v0 = 0.0f, v1 = 0.0f;
                if (g < M) {
                    const float* src = (g < splitRows) ? (A0 + (size_t)g * K)
                                                       : (A1 + (size_t)(g - splitRows) * K);
                    if (k0     < K) v0 = src[k0];
                    if (k0 + 1 < K) v1 = src[k0 + 1];
                }
                alds[idx] = pack2(v0, v1);
            }
        }
        __syncthreads();
        bf16x16 bfr = load_b_frag(Bp + (size_t)(kt * 8 + wave) * 256);
#pragma unroll
        for (int t = 0; t < 4; ++t) {
            bf16x16 afr = load_a_frag(alds, t * 16, 16, 0);
            acc[t] = __builtin_amdgcn_wmma_f32_16x16x32_bf16(
                false, afr, false, bfr, (short)0, acc[t], false, false);
        }
    }
    int n = wave * 16 + (lane & 15);
    int mh = 8 * (lane >> 4);
#pragma unroll
    for (int t = 0; t < 4; ++t)
#pragma unroll
        for (int vg = 0; vg < 8; ++vg) {
            int m = rowb + t * 16 + mh + vg;
            if (m < M) C[(size_t)m * 128 + n] = acc[t][vg];
        }
}

// ---------- hl/hr: per-node dot(h, al), dot(h, ar) ----------
__global__ void hlhr_kernel(const float* __restrict__ h, const float* __restrict__ al,
                            const float* __restrict__ ar, float* hl, float* hr, int Nn) {
    int w = (blockIdx.x * blockDim.x + threadIdx.x) >> 5;
    int lane = threadIdx.x & 31;
    if (w >= Nn) return;
    const float* row = h + (size_t)w * DD;
    float a = 0.0f, b = 0.0f;
    for (int i = lane; i < DD; i += 32) { float v = row[i]; a += v * al[i]; b += v * ar[i]; }
    for (int off = 16; off; off >>= 1) { a += __shfl_down(a, off, 32); b += __shfl_down(b, off, 32); }
    if (lane == 0) { hl[w] = a; hr[w] = b; }
}

// ---------- edge pass 1: e = leaky_relu(hl[src]+hr[dst]); segment max over dst ----------
__global__ void edge_max_kernel(const int* __restrict__ src, const int* __restrict__ dst,
                                const float* __restrict__ hl, const float* __restrict__ hr,
                                float* __restrict__ e_buf, float* __restrict__ mmax) {
    int i = blockIdx.x * blockDim.x + threadIdx.x;
    if (i >= EE) return;
    int d = dst[i];
    if (d >= DXV_) return;                       // only dst < DXV feeds the output
    float e = hl[src[i]] + hr[d];
    e = (e > 0.0f) ? e : NEG_SLOPE * e;
    e_buf[i] = e;
    atomicMaxFloat(&mmax[d], e);
}

// ---------- edge pass 2: ex = exp(e - m[dst]); denom[dst] += ex ----------
__global__ void edge_exp_kernel(const int* __restrict__ dst, float* __restrict__ e_buf,
                                const float* __restrict__ mmax, float* __restrict__ denom) {
    int i = blockIdx.x * blockDim.x + threadIdx.x;
    if (i >= EE) return;
    int d = dst[i];
    if (d >= DXV_) return;
    float ex = __expf(e_buf[i] - mmax[d]);
    e_buf[i] = ex;
    atomicAdd(&denom[d], ex);
}

// ---------- edge pass 3: gat[dst] += (ex/denom[dst]) * h[src]  (one wave per edge) ----------
__global__ void edge_scatter_kernel(const int* __restrict__ src, const int* __restrict__ dst,
                                    const float* __restrict__ e_buf, const float* __restrict__ denom,
                                    const float* __restrict__ h, float* __restrict__ gat) {
    int ew = (blockIdx.x * blockDim.x + threadIdx.x) >> 5;
    int lane = threadIdx.x & 31;
    if (ew >= EE) return;
    int d = dst[ew];
    if (d >= DXV_) return;
    float alpha = e_buf[ew] / denom[d];
    const float* hs = h + (size_t)src[ew] * DD;
    float* go = gat + (size_t)d * DD;
    for (int i = lane; i < DD; i += 32) atomicAdd(&go[i], alpha * hs[i]);
}

// ---------- fuse: dxALLonto[r] = gat[r] + c[r]/(||c[r]||+1e-12); zero row DXV ----------
__global__ void fuse_kernel(const float* __restrict__ gat, const float* __restrict__ c,
                            float* __restrict__ onto) {
    int w = (blockIdx.x * blockDim.x + threadIdx.x) >> 5;
    int lane = threadIdx.x & 31;
    if (w > DXV_) return;
    float* dst = onto + (size_t)w * DD;
    if (w == DXV_) { for (int i = lane; i < DD; i += 32) dst[i] = 0.0f; return; }
    const float* cr = c + (size_t)w * DD;
    float ss = 0.0f;
    for (int i = lane; i < DD; i += 32) { float v = cr[i]; ss += v * v; }
    for (int off = 16; off; off >>= 1) ss += __shfl_down(ss, off, 32);
    ss = __shfl(ss, 0, 32);
    float rn = 1.0f / (sqrtf(ss) + 1e-12f);
    const float* gr = gat + (size_t)w * DD;
    for (int i = lane; i < DD; i += 32) dst[i] = gr[i] + cr[i] * rn;
}

// ---------- per-visit attention: gather codes, attn GEMM (WMMA), softmax, weighted sum ----------
// a_in = [codes | u] and u is identical for every row, so the K=256..511 half of the
// GEMM yields one tile per wave: compute it once (broadcast-u A fragment) and use it as
// the initial accumulator for all 5 M-tiles.  48 WMMAs/wave instead of 80.
__global__ void visit_kernel(const int* __restrict__ dxseqs, const int* __restrict__ drugseqs,
                             const float* __restrict__ dxEmb, const float* __restrict__ onto,
                             const float* __restrict__ drugEmb, const float* __restrict__ drugEmb2,
                             const unsigned* __restrict__ Bp_attn, const float* __restrict__ attn_b,
                             const float* __restrict__ comb_W, const float* __restrict__ comb_b,
                             float* __restrict__ visit) {
    __shared__ unsigned codes[80 * 128];    // 80 codes x 256 dims as bf16 pairs (40KB)
    __shared__ unsigned upair[128];         // mean vector as bf16 pairs
    __shared__ float scores[80];
    __shared__ float red0;
    int bv = blockIdx.x;
    int tid = threadIdx.x;
    const int* dxrow = dxseqs + (size_t)bv * NDC;
    const int* drrow = drugseqs + (size_t)bv * NRC;

    // gather codes: j<40 -> [EHRdxEmb | dxALLonto], j>=40 -> [EHRdrugEmb | EHRdrugEmb2]
    for (int idx = tid; idx < 80 * 128; idx += 256) {
        int j = idx >> 7, jp = idx & 127;
        int d0 = jp * 2;
        const float* src;
        if (j < 40) {
            int id = dxrow[j];
            src = (d0 < DD) ? (dxEmb + (size_t)id * DD + d0) : (onto + (size_t)id * DD + (d0 - DD));
        } else {
            int id = drrow[j - 40];
            src = (d0 < DD) ? (drugEmb + (size_t)id * DD + d0) : (drugEmb2 + (size_t)id * DD + (d0 - DD));
        }
        const float2 v = *(const float2*)src;
        codes[idx] = pack2(v.x, v.y);
    }
    __syncthreads();
    // u = mean over the 80 codes
    if (tid < 128) {
        float s0 = 0.0f, s1 = 0.0f;
        for (int j = 0; j < 80; ++j) {
            unsigned p = codes[j * 128 + tid];
            s0 += bfbits2f(p & 0xffffu); s1 += bfbits2f(p >> 16);
        }
        upair[tid] = pack2(s0 * (1.0f / 80.0f), s1 * (1.0f / 80.0f));
    }
    if (tid < 80) scores[tid] = comb_b[0];
    __syncthreads();

    int wave = tid >> 5, lane = tid & 31;
    int n = wave * 16 + (lane & 15);
    float bn = attn_b[n], cwn = comb_W[n];
    const int jb = (lane >> 4) ? 4 : 0;

    // u-part tile (kt = 8..15 of the logical K=512 loop), broadcast across rows
    f32x8 accu;
#pragma unroll
    for (int i = 0; i < 8; ++i) accu[i] = 0.0f;
#pragma unroll 1
    for (int kt = 0; kt < 8; ++kt) {
        union { bf16x16 v; unsigned u[8]; } afr;
        const unsigned* up = upair + kt * 16 + jb;
#pragma unroll
        for (int i = 0; i < 4; ++i) afr.u[i] = up[i];
#pragma unroll
        for (int i = 0; i < 4; ++i) afr.u[4 + i] = up[8 + i];
        bf16x16 bfr = load_b_frag(Bp_attn + (size_t)((8 + kt) * 8 + wave) * 256);
        accu = __builtin_amdgcn_wmma_f32_16x16x32_bf16(
            false, afr.v, false, bfr, (short)0, accu, false, false);
    }

#pragma unroll 1
    for (int mt = 0; mt < 5; ++mt) {
        f32x8 acc = accu;
        for (int kt = 0; kt < 8; ++kt) {
            bf16x16 afr = load_a_frag(codes, mt * 16, 128, kt * 16);
            bf16x16 bfr = load_b_frag(Bp_attn + (size_t)(kt * 8 + wave) * 256);
            acc = __builtin_amdgcn_wmma_f32_16x16x32_bf16(
                false, afr, false, bfr, (short)0, acc, false, false);
        }
        int mh = 8 * (lane >> 4);
#pragma unroll
        for (int vg = 0; vg < 8; ++vg) {
            int m = mt * 16 + mh + vg;
            float s = tanhf(acc[vg] + bn) * cwn;   // tanh -> combine weight, reduce over N
            atomicAdd(&scores[m], s);
        }
    }
    __syncthreads();
    if (tid == 0) {
        float mx = -1e30f;
        for (int j = 0; j < 80; ++j) mx = fmaxf(mx, scores[j]);
        float sum = 0.0f;
        for (int j = 0; j < 80; ++j) { float e = __expf(scores[j] - mx); scores[j] = e; sum += e; }
        red0 = 1.0f / sum;
    }
    __syncthreads();
    float rs = red0;
    // visit[d] = sum_j alpha_j * codes[j][d]
    int d = tid;                 // 0..255
    float v = 0.0f;
    for (int j = 0; j < 80; ++j) {
        unsigned p = codes[j * 128 + (d >> 1)];
        v += scores[j] * bfbits2f((d & 1) ? (p >> 16) : (p & 0xffffu));
    }
    visit[(size_t)bv * 256 + d] = v * rs;
}

// ---------- final: out = sigmoid(visit[640,256] @ dp_W[256,2000] + dp_b) ----------
__global__ void gemm_out_kernel(const float* __restrict__ Av, const unsigned* __restrict__ Bp,
                                const float* __restrict__ bias, float* __restrict__ out) {
    __shared__ unsigned alds[64 * 128];     // 64 rows x 256 dims (32KB)
    int tid = threadIdx.x, wave = tid >> 5, lane = tid & 31;
    int rowb = blockIdx.x * 64;
    int ntile = blockIdx.y * 8 + wave;      // 0..127 (125 real)
    for (int idx = tid; idx < 64 * 128; idx += 256) {
        int r = idx >> 7, jp = idx & 127;
        const float2 v = *(const float2*)(Av + (size_t)(rowb + r) * 256 + jp * 2);
        alds[idx] = pack2(v.x, v.y);
    }
    __syncthreads();
    f32x8 acc[4];
#pragma unroll
    for (int t = 0; t < 4; ++t)
#pragma unroll
        for (int i = 0; i < 8; ++i) acc[t][i] = 0.0f;
#pragma unroll 1
    for (int kt = 0; kt < 8; ++kt) {
        bf16x16 bfr = load_b_frag(Bp + (size_t)(kt * 128 + ntile) * 256);
#pragma unroll
        for (int t = 0; t < 4; ++t) {
            bf16x16 afr = load_a_frag(alds, t * 16, 128, kt * 16);
            acc[t] = __builtin_amdgcn_wmma_f32_16x16x32_bf16(
                false, afr, false, bfr, (short)0, acc[t], false, false);
        }
    }
    int n = ntile * 16 + (lane & 15);
    if (n >= LBL_) return;
    float bn = bias[n];
    int mh = 8 * (lane >> 4);
#pragma unroll
    for (int t = 0; t < 4; ++t)
#pragma unroll
        for (int vg = 0; vg < 8; ++vg) {
            int m = rowb + t * 16 + mh + vg;
            float v = acc[t][vg] + bn;
            out[(size_t)m * LBL_ + n] = 1.0f / (1.0f + __expf(-v));
        }
}

extern "C" void kernel_launch(void* const* d_in, const int* in_sizes, int n_in,
                              void* d_out, int out_size, void* d_ws, size_t ws_size,
                              hipStream_t stream) {
    (void)in_sizes; (void)n_in; (void)out_size; (void)ws_size;
    const int*   dxseqs   = (const int*)d_in[0];
    const int*   drugseqs = (const int*)d_in[1];
    const int*   esrc     = (const int*)d_in[2];
    const int*   edst     = (const int*)d_in[3];
    const float* dx2anc   = (const float*)d_in[4];
    const float* dxEmb    = (const float*)d_in[5];
    const float* drugEmb  = (const float*)d_in[6];
    const float* drugEmb2 = (const float*)d_in[7];
    const float* extra_dx = (const float*)d_in[8];
    const float* singleEmb= (const float*)d_in[9];
    const float* gat_W    = (const float*)d_in[10];
    const float* gat_al   = (const float*)d_in[11];
    const float* gat_ar   = (const float*)d_in[12];
    const float* attn_W   = (const float*)d_in[13];
    const float* attn_b   = (const float*)d_in[14];
    const float* comb_W   = (const float*)d_in[15];
    const float* comb_b   = (const float*)d_in[16];
    const float* dp_W     = (const float*)d_in[17];
    const float* dp_b     = (const float*)d_in[18];
    float* out = (float*)d_out;

    char* base = (char*)d_ws;
    size_t off = 0;
    auto A = [&](size_t bytes) -> char* {
        char* p = base + off; off += (bytes + 255) & ~(size_t)255; return p;
    };
    float* c      = (float*)A((size_t)DXV_ * DD * 4);
    float* h      = (float*)A((size_t)NN_ * DD * 4);
    float* hl     = (float*)A((size_t)NN_ * 4);
    float* hr     = (float*)A((size_t)NN_ * 4);
    float* e_buf  = (float*)A((size_t)EE * 4);
    float* mmax   = (float*)A((size_t)DXV_ * 4);
    float* denom  = (float*)A((size_t)DXV_ * 4);
    float* gat    = (float*)A((size_t)DXV_ * DD * 4);
    float* onto   = (float*)A((size_t)(DXV_ + 1) * DD * 4);
    float* visit  = (float*)A((size_t)BB * VV * 2 * DD * 4);
    unsigned* bp_single = (unsigned*)A((size_t)63 * 8 * 256 * 4);
    unsigned* bp_gat    = (unsigned*)A((size_t)4 * 8 * 256 * 4);
    unsigned* bp_attn   = (unsigned*)A((size_t)16 * 8 * 256 * 4);
    unsigned* bp_dp     = (unsigned*)A((size_t)8 * 128 * 256 * 4);

    // pack all B matrices into WMMA fragment layout (bf16)
    pack_b_kernel<<<63 * 8, 32, 0, stream>>>(singleEmb, bp_single, NANC_, DD, 8);
    pack_b_kernel<<<4 * 8, 32, 0, stream>>>(gat_W, bp_gat, DD, DD, 8);
    pack_b_kernel<<<16 * 8, 32, 0, stream>>>(attn_W, bp_attn, 4 * DD, PATTN_, 8);
    pack_b_kernel<<<8 * 128, 32, 0, stream>>>(dp_W, bp_dp, 2 * DD, LBL_, 128);
    init_kernel<<<(DXV_ * DD + 255) / 256, 256, 0, stream>>>(gat, mmax, denom);

    // c = dx_to_ancestor @ SingleclassEmb   [10000,2000]x[2000,128]
    gemm128_kernel<<<(DXV_ + 63) / 64, 256, 0, stream>>>(dx2anc, dx2anc, DXV_, bp_single,
                                                         c, DXV_, NANC_, 63);
    // h = nodes @ gat_W   nodes = [EHRdxEmb[:DXV] ; extra_dx]
    gemm128_kernel<<<(NN_ + 63) / 64, 256, 0, stream>>>(dxEmb, extra_dx, DXV_, bp_gat,
                                                        h, NN_, DD, 4);
    hlhr_kernel<<<(NN_ * 32 + 255) / 256, 256, 0, stream>>>(h, gat_al, gat_ar, hl, hr, NN_);
    edge_max_kernel<<<(EE + 255) / 256, 256, 0, stream>>>(esrc, edst, hl, hr, e_buf, mmax);
    edge_exp_kernel<<<(EE + 255) / 256, 256, 0, stream>>>(edst, e_buf, mmax, denom);
    edge_scatter_kernel<<<((size_t)EE * 32 + 255) / 256, 256, 0, stream>>>(esrc, edst, e_buf,
                                                                           denom, h, gat);
    fuse_kernel<<<((DXV_ + 1) * 32 + 255) / 256, 256, 0, stream>>>(gat, c, onto);

    visit_kernel<<<BB * VV, 256, 0, stream>>>(dxseqs, drugseqs, dxEmb, onto, drugEmb, drugEmb2,
                                              bp_attn, attn_b, comb_W, comb_b, visit);
    dim3 og(10, 16);
    gemm_out_kernel<<<og, 256, 0, stream>>>(visit, bp_dp, dp_b, out);
}
